// MLA_13357348291384
// MI455X (gfx1250) — matrix operand order repaired and checked by reading, not compiled, not run on hardware
//
#include <hip/hip_runtime.h>
#include <hip/hip_bf16.h>
#include <math.h>

// ---------------------------------------------------------------------------
// MLA forward for MI455X (gfx1250): all GEMMs + attention via
// v_wmma_f32_16x16x32_bf16 (wave32). Weights pre-packed bf16 transposed;
// GEMM and attention tiles staged with double-buffered async global->LDS
// copies (ASYNCcnt partial waits) so DMA overlaps WMMA. fp32 accumulation.
// ---------------------------------------------------------------------------

#define BATCH 2
#define SEQ   2048
#define DMODEL 2048
#define NHEAD 16
#define QLORA 1536
#define KVLORA 512
#define DNOPE 128
#define DROPE 64
#define DV    128
#define DQK   192            // DNOPE + DROPE
#define BS    (BATCH * SEQ)  // 4096

typedef __attribute__((ext_vector_type(16))) __bf16 v16bf;
typedef __attribute__((ext_vector_type(8)))  float  v8f;

union FragB16 { v16bf v; uint4 q[2]; };
union Pack8   { __hip_bfloat16 h[8]; uint4 q; };

__device__ inline v16bf load_frag(const __hip_bfloat16* p0, const __hip_bfloat16* p1) {
    FragB16 f;
    f.q[0] = *reinterpret_cast<const uint4*>(p0);
    f.q[1] = *reinterpret_cast<const uint4*>(p1);
    return f.v;
}

__device__ inline v8f vzero8() {
    v8f z;
#pragma unroll
    for (int i = 0; i < 8; ++i) z[i] = 0.0f;
    return z;
}

// CDNA5 async global->LDS copy (16B per lane), tracked by ASYNCcnt.
__device__ inline void async_copy_b128(void* lds_ptr, const void* gptr) {
    uint32_t lds_off = (uint32_t)(uintptr_t)lds_ptr;   // low 32 bits = LDS offset
    asm volatile("global_load_async_to_lds_b128 %0, %1, off"
                 :: "v"(lds_off), "v"((unsigned long long)(uintptr_t)gptr)
                 : "memory");
}
__device__ inline void async_wait0() {
    asm volatile("s_wait_asynccnt 0x0" ::: "memory");
}
__device__ inline void async_wait3() {
    asm volatile("s_wait_asynccnt 0x3" ::: "memory");
}
__device__ inline void async_wait4() {
    asm volatile("s_wait_asynccnt 0x4" ::: "memory");
}

__device__ inline float redmax16(float v) {
#pragma unroll
    for (int off = 8; off >= 1; off >>= 1) v = fmaxf(v, __shfl_xor(v, off, 32));
    return v;
}
__device__ inline float redsum16(float v) {
#pragma unroll
    for (int off = 8; off >= 1; off >>= 1) v += __shfl_xor(v, off, 32);
    return v;
}

// ---------------------------------------------------------------------------
// GEMM: C[M,N](f32) = A[M,K](bf16, row-major) * W[N,K](bf16, pre-transposed)
// Block tile 128x128, K-step 32, 256 threads = 8 waves, wave tile 32x64.
// Both tiles staged via async b128 copies, double-buffered; steady-state
// loop is branch-free (stage + s_wait_asynccnt 0x4), last tile peeled.
// ---------------------------------------------------------------------------
#define GP 40  // LDS row pitch in bf16 elements (80 bytes, 16B multiple)

__global__ __launch_bounds__(256) void mla_gemm_bf16(
    const __hip_bfloat16* __restrict__ A, const __hip_bfloat16* __restrict__ Wt,
    float* __restrict__ C, int M, int N, int K)
{
    __shared__ __align__(16) __hip_bfloat16 As[2][128][GP];
    __shared__ __align__(16) __hip_bfloat16 Bs[2][128][GP];   // [n][k]

    const int t    = threadIdx.x;
    const int lane = t & 31;
    const int wid  = t >> 5;
    const int wm   = wid & 3;        // wave row group (32 rows)
    const int wn   = wid >> 2;       // wave col group (64 cols)
    const int rowBase = blockIdx.x * 128;
    const int colBase = blockIdx.y * 128;
    const int hi  = lane >> 4;       // 0: lanes 0-15, 1: lanes 16-31
    const int l15 = lane & 15;

    v8f acc[2][4];
#pragma unroll
    for (int i = 0; i < 2; ++i)
#pragma unroll
        for (int j = 0; j < 4; ++j) acc[i][j] = vzero8();

    // stage one 128x32 A tile + 128x32 B tile: 4 async b128 ops per thread
    auto stage = [&](int buf, int kt) {
#pragma unroll
        for (int i = 0; i < 2; ++i) {
            int c = t + i * 256;                 // 0..511
            int r = c >> 2, c4 = c & 3;          // chunk = 8 bf16
            async_copy_b128(&As[buf][r][c4 * 8],
                            &A[(size_t)(rowBase + r) * K + kt + c4 * 8]);
        }
#pragma unroll
        for (int i = 0; i < 2; ++i) {
            int c = t + i * 256;
            int n = c >> 2, c4 = c & 3;
            async_copy_b128(&Bs[buf][n][c4 * 8],
                            &Wt[(size_t)(colBase + n) * K + kt + c4 * 8]);
        }
    };

    auto compute = [&](int buf) {
        v16bf af[2], bfv[4];
#pragma unroll
        for (int mi = 0; mi < 2; ++mi) {
            int r = wm * 32 + mi * 16 + l15;
            af[mi] = load_frag(&As[buf][r][hi * 8], &As[buf][r][16 + hi * 8]);
        }
#pragma unroll
        for (int ni = 0; ni < 4; ++ni) {
            int n = wn * 64 + ni * 16 + l15;
            bfv[ni] = load_frag(&Bs[buf][n][hi * 8], &Bs[buf][n][16 + hi * 8]);
        }
#pragma unroll
        for (int mi = 0; mi < 2; ++mi)
#pragma unroll
            for (int ni = 0; ni < 4; ++ni)
                acc[mi][ni] = __builtin_amdgcn_wmma_f32_16x16x32_bf16(
                    false, af[mi], false, bfv[ni], (short)0, acc[mi][ni], false, false);
    };

    stage(0, 0);
    int cur = 0;
    for (int kt = 0; kt < K - 32; kt += 32) {     // steady state: branch-free
        stage(cur ^ 1, kt + 32);                  // prefetch next tile (async)
        async_wait4();                            // previous tile's 4 ops done
        __syncthreads();
        compute(cur);
        __syncthreads();
        cur ^= 1;
    }
    async_wait0();                                // drain last tile
    __syncthreads();
    compute(cur);

#pragma unroll
    for (int mi = 0; mi < 2; ++mi)
#pragma unroll
        for (int ni = 0; ni < 4; ++ni) {
            int col = colBase + wn * 64 + ni * 16 + l15;
            if (col < N) {
#pragma unroll
                for (int r = 0; r < 8; ++r) {
                    int row = rowBase + wm * 32 + mi * 16 + r + hi * 8;
                    C[(size_t)row * N + col] = acc[mi][ni][r];
                }
            }
        }
}

// ---------------------------------------------------------------------------
// Flash attention: one block = (b, h, 128 q rows). 8 waves x 16 q rows each.
// Streams 32-key tiles (double-buffered: async K copies + packed V transpose);
// online softmax; all matmuls via bf16 WMMA.
// ---------------------------------------------------------------------------
__global__ __launch_bounds__(256) void mla_attn(
    const __hip_bfloat16* __restrict__ Qb,  // [B,H,S,192]
    const __hip_bfloat16* __restrict__ Kb,  // [B,H,S,192]
    const __hip_bfloat16* __restrict__ Vb,  // [B,H,S,128]
    __hip_bfloat16* __restrict__ Ob,        // [B*S, H*128]
    float scale)
{
    __shared__ __align__(16) __hip_bfloat16 Qs[128][DQK];      // 48 KB
    __shared__ __align__(16) __hip_bfloat16 Ks[2][32][DQK];    // 24 KB
    __shared__ __align__(16) __hip_bfloat16 VTs[2][128][32];   // 16 KB (V^T)
    __shared__ __align__(16) __hip_bfloat16 Ps[8][16][32];     //  8 KB (wave P)

    const int t = threadIdx.x, lane = t & 31, wid = t >> 5;
    const int hi = lane >> 4, l15 = lane & 15;
    const int qbase = blockIdx.x * 128;
    const int h = blockIdx.y, b = blockIdx.z;
    const size_t qk_base = (size_t)(b * NHEAD + h) * SEQ * DQK;
    const size_t v_base  = (size_t)(b * NHEAD + h) * SEQ * DV;

    // stage Q (pre-multiplied by softmax scale), 8 bf16 per access
    for (int i = t; i < 128 * (DQK / 8); i += 256) {
        int r = i / (DQK / 8), c8 = i % (DQK / 8);
        Pack8 in, outp;
        in.q = *reinterpret_cast<const uint4*>(
            &Qb[qk_base + (size_t)(qbase + r) * DQK + c8 * 8]);
#pragma unroll
        for (int j = 0; j < 8; ++j)
            outp.h[j] = __float2bfloat16(__bfloat162float(in.h[j]) * scale);
        *reinterpret_cast<uint4*>(&Qs[r][c8 * 8]) = outp.q;
    }

    v8f Oacc[8];
#pragma unroll
    for (int n = 0; n < 8; ++n) Oacc[n] = vzero8();
    float mrow[8], lrow[8];
#pragma unroll
    for (int r = 0; r < 8; ++r) { mrow[r] = -INFINITY; lrow[r] = 0.0f; }

    // stage one 32-key tile: 3 async K copies + packed V transpose per thread
    auto stage_kv = [&](int buf, int kt) {
        const int kb0 = kt * 32;
#pragma unroll
        for (int i = 0; i < 3; ++i) {
            int c = t + i * 256;
            int r = c / (DQK / 8), c8 = c % (DQK / 8);
            async_copy_b128(&Ks[buf][r][c8 * 8],
                            &Kb[qk_base + (size_t)(kb0 + r) * DQK + c8 * 8]);
        }
#pragma unroll
        for (int i = 0; i < 2; ++i) {
            int task = t + i * 256;
            int d = task & 127, kg = task >> 7;
            Pack8 p;
#pragma unroll
            for (int j = 0; j < 8; ++j)
                p.h[j] = Vb[v_base + (size_t)(kb0 + kg * 8 + j) * DV + d];
            *reinterpret_cast<uint4*>(&VTs[buf][d][kg * 8]) = p.q;
        }
    };

    auto tile_compute = [&](int kt, int buf) {
        const int kb0 = kt * 32;
        // scores: 16 q rows x 32 keys = 2 accumulators, K-dim = 192 (6 steps)
        v8f sc[2];
        sc[0] = vzero8(); sc[1] = vzero8();
        const int qrow = wid * 16 + l15;
#pragma unroll
        for (int dk = 0; dk < 6; ++dk) {
            v16bf aq = load_frag(&Qs[qrow][dk * 32 + hi * 8],
                                 &Qs[qrow][dk * 32 + 16 + hi * 8]);
#pragma unroll
            for (int j = 0; j < 2; ++j) {
                v16bf bk = load_frag(&Ks[buf][j * 16 + l15][dk * 32 + hi * 8],
                                     &Ks[buf][j * 16 + l15][dk * 32 + 16 + hi * 8]);
                sc[j] = __builtin_amdgcn_wmma_f32_16x16x32_bf16(
                    false, aq, false, bk, (short)0, sc[j], false, false);
            }
        }

        // causal mask + online softmax (rows live in 16-lane halves)
        float alpha[8];
#pragma unroll
        for (int r = 0; r < 8; ++r) {
            int rowg = qbase + wid * 16 + r + hi * 8;
            int key0 = kb0 + l15, key1 = kb0 + 16 + l15;
            float s0 = (key0 <= rowg) ? sc[0][r] : -INFINITY;
            float s1 = (key1 <= rowg) ? sc[1][r] : -INFINITY;
            float mnew = fmaxf(mrow[r], redmax16(fmaxf(s0, s1)));
            float a  = __expf(mrow[r] - mnew);
            float p0 = __expf(s0 - mnew);
            float p1 = __expf(s1 - mnew);
            lrow[r] = lrow[r] * a + redsum16(p0 + p1);
            mrow[r] = mnew;
            alpha[r] = a;
            Ps[wid][r + hi * 8][l15]      = __float2bfloat16(p0);
            Ps[wid][r + hi * 8][16 + l15] = __float2bfloat16(p1);
        }

        // rescale running output, then O += P @ V
#pragma unroll
        for (int n = 0; n < 8; ++n)
#pragma unroll
            for (int r = 0; r < 8; ++r) Oacc[n][r] *= alpha[r];

        v16bf ap = load_frag(&Ps[wid][l15][hi * 8], &Ps[wid][l15][16 + hi * 8]);
#pragma unroll
        for (int n = 0; n < 8; ++n) {
            v16bf bv = load_frag(&VTs[buf][n * 16 + l15][hi * 8],
                                 &VTs[buf][n * 16 + l15][16 + hi * 8]);
            Oacc[n] = __builtin_amdgcn_wmma_f32_16x16x32_bf16(
                false, ap, false, bv, (short)0, Oacc[n], false, false);
        }
    };

    const int nkt = qbase / 32 + 4;   // causal: keys up to qbase+127
    stage_kv(0, 0);
    int cur = 0;
    for (int kt = 0; kt < nkt - 1; ++kt) {        // steady state: branch-free
        stage_kv(cur ^ 1, kt + 1);                // prefetch next key tile
        async_wait3();                            // previous tile's 3 ops done
        __syncthreads();
        tile_compute(kt, cur);
        __syncthreads();
        cur ^= 1;
    }
    async_wait0();                                // drain last tile
    __syncthreads();
    tile_compute(nkt - 1, cur);

    float rinv[8];
#pragma unroll
    for (int r = 0; r < 8; ++r) rinv[r] = 1.0f / lrow[r];
#pragma unroll
    for (int n = 0; n < 8; ++n)
#pragma unroll
        for (int r = 0; r < 8; ++r) {
            int rowg = qbase + wid * 16 + r + hi * 8;
            Ob[(size_t)(b * SEQ + rowg) * (NHEAD * DV) + h * DV + n * 16 + l15] =
                __float2bfloat16(Oacc[n][r] * rinv[r]);
        }
}

// ---------------------------------------------------------------------------
// Weight pre-pack: W[K,N] f32 -> Wt[Npad,K] bf16 (transposed, zero-padded)
// ---------------------------------------------------------------------------
__global__ __launch_bounds__(256) void mla_w_to_bf16t(
    const float* __restrict__ W, __hip_bfloat16* __restrict__ Wt,
    int K, int N, int Npad)
{
    const size_t total = (size_t)K * Npad;
    for (size_t idx = (size_t)blockIdx.x * 256 + threadIdx.x; idx < total;
         idx += (size_t)gridDim.x * 256) {
        int k = (int)(idx / Npad);
        int n = (int)(idx % Npad);
        float v = (n < N) ? W[(size_t)k * N + n] : 0.0f;
        Wt[(size_t)n * K + k] = __float2bfloat16(v);
    }
}

// ---------------------------------------------------------------------------
// Elementwise / norm / rope helper kernels
// ---------------------------------------------------------------------------
__global__ __launch_bounds__(256) void mla_f32_to_bf16(
    const float* __restrict__ X, __hip_bfloat16* __restrict__ Y, size_t n)
{
    for (size_t i = (size_t)blockIdx.x * 256 + threadIdx.x; i < n;
         i += (size_t)gridDim.x * 256)
        Y[i] = __float2bfloat16(X[i]);
}

__global__ __launch_bounds__(256) void mla_rmsnorm(
    const float* __restrict__ X, const float* __restrict__ W,
    __hip_bfloat16* __restrict__ Y, int ncol)
{
    __shared__ float red[256];
    const int row = blockIdx.x, t = threadIdx.x;
    const float* xr = X + (size_t)row * ncol;
    float s = 0.0f;
    for (int i = t; i < ncol; i += 256) { float v = xr[i]; s += v * v; }
    red[t] = s; __syncthreads();
    for (int off = 128; off >= 1; off >>= 1) {
        if (t < off) red[t] += red[t + off];
        __syncthreads();
    }
    float rn = rsqrtf(red[0] / (float)ncol + 1e-6f);
    for (int i = t; i < ncol; i += 256)
        Y[(size_t)row * ncol + i] = __float2bfloat16(xr[i] * rn * W[i]);
}

// kv_a[tok][0:512] -> rmsnorm -> c_kv ; kv_a[tok][512:576] -> rope -> Kb[...,128:192]
__global__ __launch_bounds__(256) void mla_kvsplit(
    const float* __restrict__ KVA, const float* __restrict__ W,
    const float* __restrict__ fcos, const float* __restrict__ fsin,
    __hip_bfloat16* __restrict__ Ckv, __hip_bfloat16* __restrict__ Kb)
{
    __shared__ float red[256];
    __shared__ __hip_bfloat16 kpe[DROPE];
    const int tok = blockIdx.x, t = threadIdx.x;
    const int b = tok / SEQ, sidx = tok % SEQ;
    const float* xr = KVA + (size_t)tok * (KVLORA + DROPE);
    float s = 0.0f;
    for (int i = t; i < KVLORA; i += 256) { float v = xr[i]; s += v * v; }
    red[t] = s; __syncthreads();
    for (int off = 128; off >= 1; off >>= 1) {
        if (t < off) red[t] += red[t + off];
        __syncthreads();
    }
    float rn = rsqrtf(red[0] / (float)KVLORA + 1e-6f);
    for (int i = t; i < KVLORA; i += 256)
        Ckv[(size_t)tok * KVLORA + i] = __float2bfloat16(xr[i] * rn * W[i]);
    if (t < DROPE / 2) {
        float xre = xr[KVLORA + 2 * t], xim = xr[KVLORA + 2 * t + 1];
        float c = fcos[(size_t)tok * (DROPE / 2) + t];
        float sn = fsin[(size_t)tok * (DROPE / 2) + t];
        kpe[2 * t]     = __float2bfloat16(xre * c - xim * sn);
        kpe[2 * t + 1] = __float2bfloat16(xre * sn + xim * c);
    }
    __syncthreads();
    for (int i = t; i < NHEAD * DROPE; i += 256) {
        int hh = i >> 6, d = i & 63;
        Kb[((size_t)(b * NHEAD + hh) * SEQ + sidx) * DQK + DNOPE + d] = kpe[d];
    }
}

// q_full[tok][h*192 ...] fp32 -> rope on last 64 -> Qb [B,H,S,192] bf16
__global__ __launch_bounds__(256) void mla_ropeq(
    const float* __restrict__ Qf, const float* __restrict__ fcos,
    const float* __restrict__ fsin, __hip_bfloat16* __restrict__ Qb)
{
    const int tok = blockIdx.x, t = threadIdx.x;
    const int b = tok / SEQ, sidx = tok % SEQ;
    const float* q = Qf + (size_t)tok * (NHEAD * DQK);
    for (int i = t; i < NHEAD * 96; i += 256) {
        int hh = i / 96, p = i % 96;
        size_t obase = ((size_t)(b * NHEAD + hh) * SEQ + sidx) * DQK;
        if (p < 64) {
            Qb[obase + 2 * p]     = __float2bfloat16(q[hh * DQK + 2 * p]);
            Qb[obase + 2 * p + 1] = __float2bfloat16(q[hh * DQK + 2 * p + 1]);
        } else {
            int j = p - 64;
            float xre = q[hh * DQK + DNOPE + 2 * j];
            float xim = q[hh * DQK + DNOPE + 2 * j + 1];
            float c = fcos[(size_t)tok * (DROPE / 2) + j];
            float sn = fsin[(size_t)tok * (DROPE / 2) + j];
            Qb[obase + DNOPE + 2 * j]     = __float2bfloat16(xre * c - xim * sn);
            Qb[obase + DNOPE + 2 * j + 1] = __float2bfloat16(xre * sn + xim * c);
        }
    }
}

// kvb[tok][h*256+c] fp32 -> Kb[...,0:128] (k_nope) and Vb[...,0:128]
__global__ __launch_bounds__(256) void mla_splitkvb(
    const float* __restrict__ KVB, __hip_bfloat16* __restrict__ Kb,
    __hip_bfloat16* __restrict__ Vb)
{
    const size_t total = (size_t)BS * NHEAD * 256;
    for (size_t idx = (size_t)blockIdx.x * 256 + threadIdx.x; idx < total;
         idx += (size_t)gridDim.x * 256) {
        int c = (int)(idx & 255);
        size_t th = idx >> 8;
        int hh = (int)(th & 15);
        int tok = (int)(th >> 4);
        int b = tok / SEQ, sidx = tok % SEQ;
        __hip_bfloat16 v = __float2bfloat16(KVB[idx]);
        if (c < DNOPE)
            Kb[((size_t)(b * NHEAD + hh) * SEQ + sidx) * DQK + c] = v;
        else
            Vb[((size_t)(b * NHEAD + hh) * SEQ + sidx) * DV + (c - DNOPE)] = v;
    }
}

// ---------------------------------------------------------------------------
// Host orchestration
// ---------------------------------------------------------------------------
extern "C" void kernel_launch(void* const* d_in, const int* in_sizes, int n_in,
                              void* d_out, int out_size, void* d_ws, size_t ws_size,
                              hipStream_t stream)
{
    const float* x      = (const float*)d_in[0];
    const float* fcos   = (const float*)d_in[1];
    const float* fsin   = (const float*)d_in[2];
    const float* wq_a   = (const float*)d_in[3];
    const float* qnw    = (const float*)d_in[4];
    const float* wq_b   = (const float*)d_in[5];
    const float* wkv_a  = (const float*)d_in[6];
    const float* kvnw   = (const float*)d_in[7];
    const float* wkv_b  = (const float*)d_in[8];
    const float* wo     = (const float*)d_in[9];
    float* out = (float*)d_out;
    char* ws = (char*)d_ws;

    // workspace slots (lifetime-based reuse), ~191 MB total
    constexpr size_t SZ_A = (size_t)BS * 4096 * 4;          // q_a / q_full / kvb
    constexpr size_t SZ_B = (size_t)BS * DMODEL * 2;        // xb / Vb
    constexpr size_t SZ_C = (size_t)BS * 576 * 4            // kv_a
                          + (size_t)BS * QLORA * 2;         // + q_an (Ob reuses)
    constexpr size_t SZ_E = (size_t)BS * KVLORA * 2;        // c_kv
    constexpr size_t SZ_F = (size_t)BS * NHEAD * DQK * 2;   // Qb (Kb same size)
    constexpr size_t OFF_A = 0;
    constexpr size_t OFF_B = OFF_A + SZ_A;
    constexpr size_t OFF_C = OFF_B + SZ_B;
    constexpr size_t OFF_E = OFF_C + SZ_C;
    constexpr size_t OFF_F = OFF_E + SZ_E;
    constexpr size_t OFF_G = OFF_F + SZ_F;
    constexpr size_t OFF_H = OFF_G + SZ_F;                  // bf16 packed weights

    float*           q_a    = (float*)(ws + OFF_A);
    float*           q_full = (float*)(ws + OFF_A);
    float*           kvb    = (float*)(ws + OFF_A);
    __hip_bfloat16*  xb     = (__hip_bfloat16*)(ws + OFF_B);
    __hip_bfloat16*  Vb     = (__hip_bfloat16*)(ws + OFF_B);
    float*           kv_a   = (float*)(ws + OFF_C);
    __hip_bfloat16*  q_an   = (__hip_bfloat16*)(ws + OFF_C + (size_t)BS * 576 * 4);
    __hip_bfloat16*  Ob     = (__hip_bfloat16*)(ws + OFF_C);
    __hip_bfloat16*  c_kv   = (__hip_bfloat16*)(ws + OFF_E);
    __hip_bfloat16*  Qb     = (__hip_bfloat16*)(ws + OFF_F);
    __hip_bfloat16*  Kb     = (__hip_bfloat16*)(ws + OFF_G);

    // pre-packed bf16 transposed weights [Npad][K]
    __hip_bfloat16* wqat  = (__hip_bfloat16*)(ws + OFF_H);                    // [1536][2048]
    __hip_bfloat16* wkvat = wqat  + (size_t)QLORA * DMODEL;                   // [640][2048]
    __hip_bfloat16* wqbt  = wkvat + (size_t)640 * DMODEL;                     // [3072][1536]
    __hip_bfloat16* wkvbt = wqbt  + (size_t)(NHEAD * DQK) * QLORA;            // [4096][512]
    __hip_bfloat16* wot   = wkvbt + (size_t)4096 * KVLORA;                    // [2048][2048]

    const float scale = 1.0f / sqrtf((float)DQK);

    // 0. pre-pack weights (one-time, transposed + padded)
    mla_w_to_bf16t<<<2048, 256, 0, stream>>>(wq_a,  wqat,  DMODEL, QLORA, QLORA);
    mla_w_to_bf16t<<<2048, 256, 0, stream>>>(wkv_a, wkvat, DMODEL, 576,   640);
    mla_w_to_bf16t<<<2048, 256, 0, stream>>>(wq_b,  wqbt,  QLORA, NHEAD * DQK, NHEAD * DQK);
    mla_w_to_bf16t<<<2048, 256, 0, stream>>>(wkv_b, wkvbt, KVLORA, 4096, 4096);
    mla_w_to_bf16t<<<2048, 256, 0, stream>>>(wo,    wot,   NHEAD * DV, DMODEL, DMODEL);
    // 1. x -> bf16
    mla_f32_to_bf16<<<4096, 256, 0, stream>>>(x, xb, (size_t)BS * DMODEL);
    // 2. q_a = xb @ wq_a           [4096,1536]
    mla_gemm_bf16<<<dim3(BS / 128, QLORA / 128), 256, 0, stream>>>(
        xb, wqat, q_a, BS, QLORA, DMODEL);
    // 3. kv_a = xb @ wkv_a         [4096,576] (B padded to 640)
    mla_gemm_bf16<<<dim3(BS / 128, 640 / 128), 256, 0, stream>>>(
        xb, wkvat, kv_a, BS, 576, DMODEL);
    // 4. q_an = rmsnorm(q_a) * qnw (bf16)
    mla_rmsnorm<<<BS, 256, 0, stream>>>(q_a, qnw, q_an, QLORA);
    // 5. c_kv = rmsnorm(kv_a[:512]); k_pe = rope -> broadcast into Kb rope slice
    mla_kvsplit<<<BS, 256, 0, stream>>>(kv_a, kvnw, fcos, fsin, c_kv, Kb);
    // 6. q_full = q_an @ wq_b      [4096,3072]
    mla_gemm_bf16<<<dim3(BS / 128, (NHEAD * DQK) / 128), 256, 0, stream>>>(
        q_an, wqbt, q_full, BS, NHEAD * DQK, QLORA);
    // 7. rope q_pe -> Qb [B,H,S,192]
    mla_ropeq<<<BS, 256, 0, stream>>>(q_full, fcos, fsin, Qb);
    // 8. kvb = c_kv @ wkv_b        [4096,4096]
    mla_gemm_bf16<<<dim3(BS / 128, 4096 / 128), 256, 0, stream>>>(
        c_kv, wkvbt, kvb, BS, NHEAD * (DNOPE + DV), KVLORA);
    // 9. split kvb -> Kb nope slice, Vb
    mla_splitkvb<<<8192, 256, 0, stream>>>(kvb, Kb, Vb);
    // 10. flash attention -> Ob [B*S, 2048] bf16
    mla_attn<<<dim3(SEQ / 128, NHEAD, BATCH), 256, 0, stream>>>(Qb, Kb, Vb, Ob, scale);
    // 11. out = Ob @ wo            [4096,2048] fp32 -> d_out
    mla_gemm_bf16<<<dim3(BS / 128, DMODEL / 128), 256, 0, stream>>>(
        Ob, wot, out, BS, DMODEL, NHEAD * DV);
}
